// VisionMamba3D_64201171140930
// MI455X (gfx1250) — matrix operand order repaired and checked by reading, not compiled, not run on hardware
//
#include <hip/hip_runtime.h>

// ---------------- types ----------------
typedef unsigned short ushort_t;
typedef __bf16 bf16;
typedef bf16  v16bf __attribute__((ext_vector_type(16)));
typedef float v8f   __attribute__((ext_vector_type(8)));
typedef unsigned int u32x4 __attribute__((ext_vector_type(4)));

__device__ __forceinline__ ushort_t f32_to_bf16(float f) {
  union { float f; unsigned int u; } v; v.f = f;
  unsigned int r = v.u + 0x7FFFu + ((v.u >> 16) & 1u);
  return (ushort_t)(r >> 16);
}
__device__ __forceinline__ float sigmoidf_(float x) { return 1.0f / (1.0f + __expf(-x)); }

// ---------------- fp32 -> bf16 weight/activation conversion ----------------
__global__ void cvt_bf16_kernel(const float* __restrict__ in, ushort_t* __restrict__ out, long long n) {
  long long i = (long long)blockIdx.x * blockDim.x + threadIdx.x;
  if (i < n) out[i] = f32_to_bf16(in[i]);
}

// ---------------- WMMA bf16 GEMM: C[M,N] = A[M,K] @ W[N,K]^T (+bias)(+act)(+resid) ----------------
// Block = 4 waves. Each wave owns a 16Mx64N strip (4 accumulators, A-fragment reused 4x).
// All 4 waves share the same 64-wide N strip of W: the W K-slice (64x32 bf16 = 4KB) is staged
// into LDS double-buffered with global_load_async_to_lds_b128 (ASYNCcnt) and read back as
// ds_load_b128 fragments. A fragments are direct global_load_b128 + global_prefetch_b8.
// K % 32 == 0. ACT: 0=none, 1=exact GELU. OUT_BF16: bf16 store. ADD_RESID: C += resid.
union FragAB { v16bf v; u32x4 q[2]; };
union FragC  { v8f  v; float f[8]; };

template <int ACT, int OUT_BF16, int ADD_RESID>
__global__ __launch_bounds__(128) void gemm_wmma_bf16(
    const ushort_t* __restrict__ A, const ushort_t* __restrict__ W,
    const float* __restrict__ bias, const float* __restrict__ resid,
    float* __restrict__ Cf, ushort_t* __restrict__ Cb, int M, int N, int K) {
  __shared__ ushort_t wlds[2][64 * 32];            // double-buffered 64-row x 32-K bf16 slice
  const int tid  = threadIdx.x;
  const int lane = tid & 31;
  const int wave = tid >> 5;
  const int n0 = blockIdx.x * 64;
  const int m0 = blockIdx.y * 64 + wave * 16;      // may exceed M: clamped loads, masked stores

  // A fragment (16x32 bf16): lane<16 -> M=lane, K in {0..7,16..23}; lane>=16 -> K in {8..15,24..31}
  int am = m0 + (lane & 15); if (am >= M) am = M - 1;
  const int a_koff = (lane >> 4) * 8;
  const int b_koff = (lane >> 4) * 8;              // B fragment: same K split as A
  const ushort_t* __restrict__ Arow = A + (size_t)am * K;

  // Stage one 64x32 W slice into LDS buffer `buf` (each thread = one lane of 2 async ops;
  // 128 threads x 2 x 16B = 4KB).
  auto stage = [&](int k0, int buf) {
#pragma unroll
    for (int phase = 0; phase < 2; ++phase) {
      int idx = phase * 128 + tid;                 // 0..255 chunk id
      int row = idx >> 2;                          // 0..63 strip row
      int cc  = (idx & 3) * 8;                     // ushort offset within 32-wide K slice
      int wn  = n0 + row; if (wn >= N) wn = N - 1; // clamp (masked at store)
      const ushort_t* gp = W + (size_t)wn * K + k0 + cc;
      unsigned lo = (unsigned)(size_t)&wlds[buf][row * 32 + cc];
      asm volatile("global_load_async_to_lds_b128 %0, %1, off"
                   :: "v"(lo), "v"(gp) : "memory");
    }
  };

  FragC acc[4];
#pragma unroll
  for (int j = 0; j < 4; ++j)
#pragma unroll
    for (int r = 0; r < 8; ++r) acc[j].f[r] = 0.0f;

  stage(0, 0);
  asm volatile("s_wait_asynccnt 0x0" ::: "memory");
  __syncthreads();

  int buf = 0;
  for (int k0 = 0; k0 < K; k0 += 32) {
    const bool more = (k0 + 32 < K);
    if (more) stage(k0 + 32, buf ^ 1);             // overlap next fill with this slice's math

    FragAB fa;
    fa.q[0] = *(const u32x4*)(Arow + k0 + a_koff);
    fa.q[1] = *(const u32x4*)(Arow + k0 + a_koff + 16);
    if (more) __builtin_prefetch(Arow + k0 + 32 + a_koff, 0, 3);  // global_prefetch_b8

#pragma unroll
    for (int j = 0; j < 4; ++j) {                  // A-fragment reused across 4 WMMAs
      FragAB fb;
      const ushort_t* bp = &wlds[buf][(j * 16 + (lane & 15)) * 32 + b_koff];
      fb.q[0] = *(const u32x4*)(bp);               // ds_load_b128
      fb.q[1] = *(const u32x4*)(bp + 16);
      acc[j].v = __builtin_amdgcn_wmma_f32_16x16x32_bf16(false, fa.v, false, fb.v,
                                                         (short)0, acc[j].v, false, false);
    }

    if (more) asm volatile("s_wait_asynccnt 0x0" ::: "memory");   // next buffer filled
    __syncthreads();                               // everyone done reading `buf`
    buf ^= 1;
  }

  // C layout per tile: lane<16 -> rows m0+0..7, col tile+lane; lane>=16 -> rows m0+8..15
  const int srow0 = m0 + ((lane >> 4) * 8);
#pragma unroll
  for (int j = 0; j < 4; ++j) {
    const int scol = n0 + j * 16 + (lane & 15);
    if (scol >= N) continue;
    const float bval = bias ? bias[scol] : 0.0f;
#pragma unroll
    for (int r = 0; r < 8; ++r) {
      int row = srow0 + r;
      if (row >= M) break;
      float v = acc[j].f[r] + bval;
      if (ACT == 1) v = 0.5f * v * (1.0f + erff(v * 0.70710678118f));  // exact GELU
      if (ADD_RESID) v += resid[(size_t)row * N + scol];
      if (OUT_BF16) Cb[(size_t)row * N + scol] = f32_to_bf16(v);
      else          Cf[(size_t)row * N + scol] = v;
    }
  }
}

// ---------------- LayerNorm (one block per token), optional f32 and/or bf16 out ----------------
__global__ __launch_bounds__(256) void layernorm_kernel(
    const float* __restrict__ X, const float* __restrict__ g, const float* __restrict__ b,
    float* __restrict__ outF, ushort_t* __restrict__ outB, int D) {
  __shared__ float red[256];
  const int t = blockIdx.x, tid = threadIdx.x;
  const float* xr = X + (size_t)t * D;
  float s = 0.0f;
  for (int i = tid; i < D; i += 256) s += xr[i];
  red[tid] = s; __syncthreads();
  for (int o = 128; o > 0; o >>= 1) { if (tid < o) red[tid] += red[tid + o]; __syncthreads(); }
  const float mean = red[0] / D; __syncthreads();
  float v = 0.0f;
  for (int i = tid; i < D; i += 256) { float d = xr[i] - mean; v += d * d; }
  red[tid] = v; __syncthreads();
  for (int o = 128; o > 0; o >>= 1) { if (tid < o) red[tid] += red[tid + o]; __syncthreads(); }
  const float rs = rsqrtf(red[0] / D + 1e-5f);
  for (int i = tid; i < D; i += 256) {
    float y = (xr[i] - mean) * rs * g[i] + b[i];
    if (outF) outF[(size_t)t * D + i] = y;
    if (outB) outB[(size_t)t * D + i] = f32_to_bf16(y);
  }
}

// ---------------- patch embed: conv3d 1->96, k=(5,8,8), s=(5,8,8); token-major out ----------------
__global__ void patch_embed_kernel(const float* __restrict__ X, const float* __restrict__ W,
                                   const float* __restrict__ B, float* __restrict__ out) {
  int idx = blockIdx.x * blockDim.x + threadIdx.x;
  if (idx >= 4096 * 96) return;
  int co = idx % 96, tok = idx / 96;
  int tx = tok & 15, ty = (tok >> 4) & 15, tz = tok >> 8;
  const float* w = W + co * 320;
  float acc = B[co];
  for (int kz = 0; kz < 5; ++kz)
    for (int ky = 0; ky < 8; ++ky)
      for (int kx = 0; kx < 8; ++kx)
        acc += X[((size_t)(tz * 5 + kz) * 128 + (ty * 8 + ky)) * 128 + (tx * 8 + kx)]
             * w[(kz * 8 + ky) * 8 + kx];
  out[(size_t)tok * 96 + co] = acc;
}

// ---------------- depthwise causal conv (w=4) + bias + SiLU; xi = xz[:, :di] ----------------
__global__ void dwconv_silu_kernel(const float* __restrict__ xz, const float* __restrict__ cw,
                                   const float* __restrict__ cb, float* __restrict__ u,
                                   ushort_t* __restrict__ ubf, int L, int di) {
  int idx = blockIdx.x * blockDim.x + threadIdx.x;
  if (idx >= L * di) return;
  int c = idx % di, t = idx / di;
  float acc = cb[c];
#pragma unroll
  for (int k = 0; k < 4; ++k) {
    int tt = t - 3 + k;
    if (tt >= 0) acc += xz[(size_t)tt * (2 * di) + c] * cw[c * 4 + k];
  }
  float s = acc * sigmoidf_(acc);
  u[idx] = s; ubf[idx] = f32_to_bf16(s);
}

// ---------------- dt projection + softplus (K = dtr <= 48, VALU) ----------------
__global__ void dtproj_softplus_kernel(const float* __restrict__ xdbl, const float* __restrict__ dtw,
                                       const float* __restrict__ dtb, float* __restrict__ delta,
                                       int L, int di, int dtr, int ldx) {
  int idx = blockIdx.x * blockDim.x + threadIdx.x;
  if (idx >= L * di) return;
  int c = idx % di, t = idx / di;
  float acc = dtb[c];
  for (int k = 0; k < dtr; ++k) acc += xdbl[(size_t)t * ldx + k] * dtw[c * dtr + k];
  delta[idx] = (acc > 20.0f) ? acc : log1pf(__expf(acc));
}

// ---------------- selective scan: one wave per channel, 2 states/lane, shfl_xor reduce ----------------
__global__ __launch_bounds__(32) void scan_kernel(
    const float* __restrict__ u, const float* __restrict__ delta, const float* __restrict__ xdbl,
    const float* __restrict__ Alog, const float* __restrict__ Dp, float* __restrict__ y,
    int L, int di, int dtr, int ldx) {
  const int d = blockIdx.x, lane = threadIdx.x;
  const int s0 = lane, s1 = lane + 32;
  const float A0 = -__expf(Alog[d * 64 + s0]);
  const float A1 = -__expf(Alog[d * 64 + s1]);
  const float Dd = Dp[d];
  float h0 = 0.0f, h1 = 0.0f;
  for (int t = 0; t < L; ++t) {
    const float dt = delta[(size_t)t * di + d];
    const float ut = u[(size_t)t * di + d];
    const float* xr = xdbl + (size_t)t * ldx + dtr;
    const float B0 = xr[s0], B1 = xr[s1], C0 = xr[64 + s0], C1 = xr[64 + s1];
    h0 = __expf(dt * A0) * h0 + dt * B0 * ut;
    h1 = __expf(dt * A1) * h1 + dt * B1 * ut;
    float ys = h0 * C0 + h1 * C1;
#pragma unroll
    for (int o = 16; o > 0; o >>= 1) ys += __shfl_xor(ys, o, 32);
    if (lane == 0) y[(size_t)t * di + d] = ys + ut * Dd;
  }
}

// ---------------- gating: g = y * silu(z), z = xz[:, di:], bf16 out ----------------
__global__ void gate_silu_kernel(const float* __restrict__ y, const float* __restrict__ xz,
                                 ushort_t* __restrict__ g, int L, int di) {
  int idx = blockIdx.x * blockDim.x + threadIdx.x;
  if (idx >= L * di) return;
  int c = idx % di, t = idx / di;
  float z = xz[(size_t)t * (2 * di) + di + c];
  g[idx] = f32_to_bf16(y[idx] * z * sigmoidf_(z));
}

// ---------------- 2x2x2 stride-2 downsample conv (token-major in/out) ----------------
__global__ void downsample_kernel(const float* __restrict__ xin, const float* __restrict__ w,
                                  const float* __restrict__ b, float* __restrict__ xout,
                                  int n, int Ci, int Co) {
  int m = n >> 1;
  int idx = blockIdx.x * blockDim.x + threadIdx.x;
  if (idx >= m * m * m * Co) return;
  int co = idx % Co, tok = idx / Co;
  int x = tok % m, yy = (tok / m) % m, z = tok / (m * m);
  float acc = b[co];
  for (int dz = 0; dz < 2; ++dz)
    for (int dy = 0; dy < 2; ++dy)
      for (int dx = 0; dx < 2; ++dx) {
        int it = ((2 * z + dz) * n + (2 * yy + dy)) * n + (2 * x + dx);
        const float* xr = xin + (size_t)it * Ci;
        const float* wr = w + ((((size_t)co * Ci) * 2 + dz) * 2 + dy) * 2 + dx;
        for (int ci = 0; ci < Ci; ++ci) acc += xr[ci] * wr[(size_t)ci * 8];
      }
  xout[(size_t)tok * Co + co] = acc;
}

// ---------------- small dense layers (VALU) ----------------
__global__ void tok_linear_kernel(const float* __restrict__ X, const float* __restrict__ w,
                                  const float* __restrict__ b, float* __restrict__ out,
                                  int T, int N, int K) {
  int idx = blockIdx.x * blockDim.x + threadIdx.x;
  if (idx >= T * N) return;
  int n = idx % N, t = idx / N;
  float acc = b[n];
  const float* xr = X + (size_t)t * K;
  const float* wr = w + (size_t)n * K;
  for (int k = 0; k < K; ++k) acc += xr[k] * wr[k];
  out[idx] = acc;
}
__global__ void meanpool_kernel(const float* __restrict__ in, float* __restrict__ out, int T, int C) {
  int c = blockIdx.x * blockDim.x + threadIdx.x;
  if (c >= C) return;
  float s = 0.0f;
  for (int t = 0; t < T; ++t) s += in[(size_t)t * C + c];
  out[c] = s / T;
}
__global__ void linear_kernel(const float* __restrict__ x, const float* __restrict__ w,
                              const float* __restrict__ b, float* __restrict__ out,
                              int N, int K, int relu) {
  int n = blockIdx.x * blockDim.x + threadIdx.x;
  if (n >= N) return;
  float acc = b[n];
  const float* wr = w + (size_t)n * K;
  for (int k = 0; k < K; ++k) acc += x[k] * wr[k];
  out[n] = (relu && acc < 0.0f) ? 0.0f : acc;
}

// ---------------- host orchestration ----------------
static inline int cdiv(int a, int b) { return (a + b - 1) / b; }

static void run_gemm(hipStream_t st, const ushort_t* A, const float* Wf, ushort_t* wbuf,
                     const float* bias, const float* resid, float* Cf, ushort_t* Cb,
                     int M, int N, int K, int gelu_bf16_out, int add_resid) {
  long long wn = (long long)N * K;
  cvt_bf16_kernel<<<cdiv((int)wn, 256), 256, 0, st>>>(Wf, wbuf, wn);
  dim3 g(cdiv(N, 64), cdiv(M, 64)), blk(128);
  if (gelu_bf16_out)
    gemm_wmma_bf16<1, 1, 0><<<g, blk, 0, st>>>(A, wbuf, bias, resid, Cf, Cb, M, N, K);
  else if (add_resid)
    gemm_wmma_bf16<0, 0, 1><<<g, blk, 0, st>>>(A, wbuf, bias, resid, Cf, Cb, M, N, K);
  else
    gemm_wmma_bf16<0, 0, 0><<<g, blk, 0, st>>>(A, wbuf, bias, resid, Cf, Cb, M, N, K);
}

struct Blk {
  const float *Alog, *Dp, *conv_b, *conv_w, *dt_b, *dt_w, *in_w;
  const float *ln1g, *ln1b, *ln4g, *ln4b, *ln5g, *ln5b, *ln6g, *ln6b;
  const float *mb1, *mb2, *mw1, *mw2, *out_w, *x_w;
};

extern "C" void kernel_launch(void* const* d_in, const int* in_sizes, int n_in,
                              void* d_out, int out_size, void* d_ws, size_t ws_size,
                              hipStream_t stream) {
  (void)in_sizes; (void)n_in; (void)out_size; (void)ws_size;
  // --- input pytree (dict keys sorted; lists in order): x, then params leaves ---
  const float* x_in = (const float*)d_in[0];
  int idx = 1;
  auto IN = [&](void) { return (const float*)d_in[idx++]; };
  const float* bott_b = IN(); const float* bott_w = IN();
  const float *down_b[3], *down_w[3];
  for (int i = 0; i < 3; ++i) { down_b[i] = IN(); down_w[i] = IN(); }
  const float* fc1_b = IN(); const float* fc1_w = IN();
  const float* fc2_b = IN(); const float* fc2_w = IN();
  const float* patch_b = IN(); const float* patch_w = IN();
  const int DEPTHS[4] = {2, 2, 6, 2};
  Blk blk[12]; int bi = 0;
  for (int s = 0; s < 4; ++s)
    for (int j = 0; j < DEPTHS[s]; ++j) {
      Blk& B = blk[bi++];
      B.Alog = IN(); B.Dp = IN(); B.conv_b = IN(); B.conv_w = IN();
      B.dt_b = IN(); B.dt_w = IN(); B.in_w = IN();
      B.ln1g = IN(); B.ln1b = IN(); B.ln4g = IN(); B.ln4b = IN();
      B.ln5g = IN(); B.ln5b = IN(); B.ln6g = IN(); B.ln6b = IN();
      B.mb1 = IN(); B.mb2 = IN(); B.mw1 = IN(); B.mw2 = IN();
      B.out_w = IN(); B.x_w = IN();
    }

  // --- workspace bump allocator ---
  char* ws = (char*)d_ws; size_t off = 0;
  auto alloc = [&](size_t bytes) -> void* { void* p = ws + off; off = (off + bytes + 255) & ~(size_t)255; return p; };
  float*    xA    = (float*)alloc(393216 * 4);     // master activation ping
  float*    xB    = (float*)alloc(393216 * 4);     // master activation pong
  ushort_t* abuf  = (ushort_t*)alloc(393216 * 2);  // LN -> bf16 GEMM A
  float*    xz    = (float*)alloc(1572864 * 4);    // in_proj out [L, 4D]
  float*    ubuf  = (float*)alloc(786432 * 4);
  ushort_t* ubf   = (ushort_t*)alloc(786432 * 2);
  float*    xdbl  = (float*)alloc(786432 * 4);     // x_proj out [L, dtr+128]
  float*    dbuf  = (float*)alloc(786432 * 4);     // delta
  float*    ybuf  = (float*)alloc(786432 * 4);
  ushort_t* gbuf  = (ushort_t*)alloc(786432 * 2);  // gated bf16
  ushort_t* hbuf  = (ushort_t*)alloc(1572864 * 2); // mlp hidden bf16
  ushort_t* wbuf  = (ushort_t*)alloc(2359296 * 2); // bf16 weight slab (max 3072x768)
  float*    bott  = (float*)alloc(8 * 768 * 4);
  float*    pooled= (float*)alloc(768 * 4);
  float*    fcbuf = (float*)alloc(512 * 4);

  const int Ls[4] = {4096, 512, 64, 8};
  const int Ds[4] = {96, 192, 384, 768};
  const int GR[4] = {16, 8, 4, 2};

  patch_embed_kernel<<<cdiv(4096 * 96, 256), 256, 0, stream>>>(x_in, patch_w, patch_b, xA);

  float* xcur = xA; float* xalt = xB;
  bi = 0;
  for (int s = 0; s < 4; ++s) {
    const int L = Ls[s], D = Ds[s], di = 2 * D, dtr = (D + 15) / 16, ldx = dtr + 128;
    for (int j = 0; j < DEPTHS[s]; ++j) {
      Blk& B = blk[bi++];
      // --- mamba branch ---
      layernorm_kernel<<<L, 256, 0, stream>>>(xcur, B.ln1g, B.ln1b, nullptr, abuf, D);
      run_gemm(stream, abuf, B.in_w, wbuf, nullptr, nullptr, xz, nullptr, L, 2 * di, D, 0, 0);
      dwconv_silu_kernel<<<cdiv(L * di, 256), 256, 0, stream>>>(xz, B.conv_w, B.conv_b, ubuf, ubf, L, di);
      run_gemm(stream, ubf, B.x_w, wbuf, nullptr, nullptr, xdbl, nullptr, L, ldx, di, 0, 0);
      dtproj_softplus_kernel<<<cdiv(L * di, 256), 256, 0, stream>>>(xdbl, B.dt_w, B.dt_b, dbuf, L, di, dtr, ldx);
      scan_kernel<<<di, 32, 0, stream>>>(ubuf, dbuf, xdbl, B.Alog, B.Dp, ybuf, L, di, dtr, ldx);
      gate_silu_kernel<<<cdiv(L * di, 256), 256, 0, stream>>>(ybuf, xz, gbuf, L, di);
      run_gemm(stream, gbuf, B.out_w, wbuf, nullptr, xcur, xcur, nullptr, L, D, di, 0, 1); // x += mamba
      // --- MLP branch ---
      layernorm_kernel<<<L, 256, 0, stream>>>(xcur, B.ln4g, B.ln4b, xcur, nullptr, D);      // x = ln4(x)
      layernorm_kernel<<<L, 256, 0, stream>>>(xcur, B.ln5g, B.ln5b, nullptr, abuf, D);      // h = ln5(x)
      run_gemm(stream, abuf, B.mw1, wbuf, B.mb1, nullptr, nullptr, hbuf, L, 4 * D, D, 1, 0); // GELU, bf16
      run_gemm(stream, hbuf, B.mw2, wbuf, B.mb2, xcur, xcur, nullptr, L, D, 4 * D, 0, 1);    // x += mlp
      layernorm_kernel<<<L, 256, 0, stream>>>(xcur, B.ln6g, B.ln6b, xcur, nullptr, D);      // x = ln6(x)
    }
    if (s < 3) {
      int m = GR[s] / 2, Co = Ds[s + 1];
      downsample_kernel<<<cdiv(m * m * m * Co, 256), 256, 0, stream>>>(xcur, down_w[s], down_b[s], xalt, GR[s], D, Co);
      float* t = xcur; xcur = xalt; xalt = t;
    }
  }

  // --- head: 1x1x1 conv, global mean pool, fc1+relu, fc2 ---
  tok_linear_kernel<<<cdiv(8 * 768, 256), 256, 0, stream>>>(xcur, bott_w, bott_b, bott, 8, 768, 768);
  meanpool_kernel<<<cdiv(768, 256), 256, 0, stream>>>(bott, pooled, 8, 768);
  linear_kernel<<<cdiv(512, 256), 256, 0, stream>>>(pooled, fc1_w, fc1_b, fcbuf, 512, 768, 1);
  linear_kernel<<<1, 32, 0, stream>>>(fcbuf, fc2_w, fc2_b, (float*)d_out, 2, 512, 0);
}
